// VTP_82781199663490
// MI455X (gfx1250) — matrix operand order repaired and checked by reading, not compiled, not run on hardware
//
#include <hip/hip_runtime.h>
#include <math.h>

// ---------------------------------------------------------------------------
// MI455X (gfx1250) implementation. wave32, WMMA bf16 (f32 accum) everywhere.
// ---------------------------------------------------------------------------

typedef __bf16 bf16;
typedef __attribute__((ext_vector_type(8)))  float v8f;
typedef __attribute__((ext_vector_type(8)))  bf16  v8bf;
typedef __attribute__((ext_vector_type(16))) bf16  v16bf;

constexpr int Bz = 16, Lz = 512, NVz = 1, Tz = 1024, Cz = 512, Hz = 8, Dz = 64;
constexpr int qN = Lz + 1;     // 513
constexpr int kN = NVz + Tz;   // 1025
constexpr int kNp = 1056;      // 33 chunks * 32 keys (padded, 32B-aligned rows)
constexpr float SCALE  = 0.125f;   // D^-0.5
constexpr float LN_EPS = 1e-5f;

constexpr size_t N_TFE = (size_t)Bz * qN * Cz;           // 4,202,496
constexpr size_t N_KVE = (size_t)Bz * kN * Cz;           // 8,396,800
constexpr size_t N_VT  = (size_t)Bz * Hz * Dz * kNp;     // 8,650,752

// workspace layout (bytes); pools reused across phases
constexpr size_t OFF_TFF  = 0;                           // tf f32        [prep..ln1]
constexpr size_t OFF_P1   = OFF_TFF + N_TFE * 4;         // tf bf16 -> attn-out bf16
constexpr size_t OFF_P2   = OFF_P1  + N_TFE * 2;         // kin bf16 -> xp f32
constexpr size_t P2_SZ    = N_TFE * 4;                   // max of the two uses
constexpr size_t OFF_VIN  = OFF_P2  + P2_SZ;             // vin bf16
constexpr size_t OFF_WBF  = OFF_VIN + N_KVE * 2;         // 5 bf16 transposed weights
constexpr size_t OFF_Q    = OFF_WBF + 5 * (size_t)Cz * Cz * 2;
constexpr size_t OFF_K    = OFF_Q   + N_TFE * 2;
constexpr size_t OFF_V    = OFF_K   + N_KVE * 2;
constexpr size_t OFF_VT   = OFF_V   + N_KVE * 2;         // transposed+interleaved V
constexpr size_t OFF_XE   = OFF_VT  + N_VT * 2;          // x_embed f32 (B*L)
constexpr size_t OFF_TXTC = OFF_XE  + (size_t)Bz * Lz * 4;  // txt_cur pre-gather
constexpr size_t OFF_TXT2 = OFF_TXTC + (size_t)Bz * Cz * 4; // txt2 bf16
constexpr size_t OFF_TXTP = OFF_TXT2 + (size_t)Bz * Cz * 2; // txt proj f32
constexpr size_t OFF_IDX  = OFF_TXTP + (size_t)Bz * Cz * 4; // argmax idx

__device__ __forceinline__ v8bf ld8(const bf16* p) {
    return *reinterpret_cast<const v8bf*>(p);
}
__device__ __forceinline__ v16bf cat16(v8bf lo, v8bf hi) {
    union { v16bf v; v8bf h[2]; } u; u.h[0] = lo; u.h[1] = hi; return u.v;
}
__device__ __forceinline__ v8f vzero8() {
    v8f z;
#pragma unroll
    for (int i = 0; i < 8; ++i) z[i] = 0.f;
    return z;
}
__device__ __forceinline__ unsigned pack2(float a, float b) {
    union { bf16 h[2]; unsigned u; } p;
    p.h[0] = (bf16)a; p.h[1] = (bf16)b;
    return p.u;
}
#define WMMA_BF16(a, b, c) \
    __builtin_amdgcn_wmma_f32_16x16x32_bf16(false, (a), false, (b), (short)0, (c), false, false)

// ---------------------------------------------------------------------------
// f32 weight [K][N] -> bf16 transposed [N][K] (all GEMM B-operands)
__global__ void k_convT(const float* __restrict__ src, bf16* __restrict__ dst) {
    int e = blockIdx.x * 256 + threadIdx.x;   // e = n*512 + k
    int n = e >> 9, k = e & 511;
    dst[e] = (bf16)src[k * Cz + n];
}

// ---------------------------------------------------------------------------
// V bf16 (b,key,C) -> vT[((b*H+h)*D+d)*kNp + chunk*32 + pos], pos interleaved:
// key = chunk*32 + j,  pos = 2*(j&15) + (j>>4).  Pad keys >= kN with zeros.
__global__ void k_vT(const bf16* __restrict__ vB, bf16* __restrict__ vT) {
    size_t e = (size_t)blockIdx.x * 256 + threadIdx.x;
    if (e >= N_VT) return;
    int pos = (int)(e % kNp);
    int d   = (int)((e / kNp) % Dz);
    int h   = (int)((e / ((size_t)kNp * Dz)) % Hz);
    int b   = (int)(e / ((size_t)kNp * Dz * Hz));
    int p   = pos & 31;
    int j   = (p >> 1) + ((p & 1) << 4);
    int key = (pos & ~31) + j;
    bf16 val = (bf16)0.f;
    if (key < kN)
        val = vB[((size_t)(b * kN + key)) * Cz + h * 64 + d];
    vT[e] = val;
}

// ---------------------------------------------------------------------------
// per-batch inclusive scan of !mask for sine positional embedding
__global__ void k_scan(const int* __restrict__ mask, float* __restrict__ xembed) {
    __shared__ float s[Lz];
    int b = blockIdx.x, t = threadIdx.x;
    s[t] = mask[b * Lz + t] ? 0.f : 1.f;
    __syncthreads();
    for (int off = 1; off < Lz; off <<= 1) {
        float add = (t >= off) ? s[t - off] : 0.f;
        __syncthreads();
        s[t] += add;
        __syncthreads();
    }
    float denom = s[Lz - 1] + 1e-6f;
    xembed[b * Lz + t] = s[t] / denom * 6.28318530717958647692f;
}

// ---------------------------------------------------------------------------
// text_features = concat(txt_token + t_pos_w, text_token + sinepos)
__global__ void k_fill_tf(const float* __restrict__ txt_tok, const float* __restrict__ text_tok,
                          const float* __restrict__ t_pos_w, const float* __restrict__ xembed,
                          float* __restrict__ tfF, bf16* __restrict__ tfB) {
    size_t e = (size_t)blockIdx.x * 256 + threadIdx.x;
    if (e >= N_TFE) return;
    int c   = (int)(e % Cz);
    int row = (int)((e / Cz) % qN);
    int b   = (int)(e / ((size_t)Cz * qN));
    float val;
    if (row == 0) {
        val = txt_tok[b * Cz + c] + t_pos_w[c];
    } else {
        int l = row - 1;
        float f = __powf(10000.0f, (float)(c & ~1) * (1.0f / Cz));
        float a = xembed[b * Lz + l] / f;
        float pos = (c & 1) ? __cosf(a) : __sinf(a);
        val = text_tok[((size_t)b * Lz + l) * Cz + c] + pos;
    }
    tfF[e] = val;
    tfB[e] = (bf16)val;
}

// ---------------------------------------------------------------------------
// k_in = concat(vis + v_pos_w, template + template_pos); v_in = concat(vis, template)
__global__ void k_fill_kv(const float* __restrict__ vis, const float* __restrict__ v_pos_w,
                          const float* __restrict__ tmpl, const float* __restrict__ tmpl_pos,
                          bf16* __restrict__ kin, bf16* __restrict__ vin) {
    size_t e = (size_t)blockIdx.x * 256 + threadIdx.x;
    if (e >= N_KVE) return;
    int c   = (int)(e % Cz);
    int row = (int)((e / Cz) % kN);
    int b   = (int)(e / ((size_t)Cz * kN));
    float kv, vv;
    if (row == 0) {
        vv = vis[b * Cz + c];
        kv = vv + v_pos_w[c];
    } else {
        size_t ti = ((size_t)b * Tz + (row - 1)) * Cz + c;
        vv = tmpl[ti];
        kv = vv + tmpl_pos[ti];
    }
    kin[e] = (bf16)kv;
    vin[e] = (bf16)vv;
}

// ---------------------------------------------------------------------------
// C[M,512] = A_bf16[M,512] @ W[512,512] + bias, W given pre-transposed [N][K].
// 128x64 block tile, 8 waves x (32x32 per wave = 4 wmma accumulators):
// 4 WMMAs per 2 A-frag + 2 B-frag LDS loads.
__global__ __launch_bounds__(256) void k_gemm_bf16(
        const bf16* __restrict__ A, const bf16* __restrict__ WT,
        const float* __restrict__ bias, float* __restrict__ outF,
        bf16* __restrict__ outB, int M) {
    constexpr int K = Cz, N = Cz;
    __shared__ __align__(16) bf16 lds_a[128 * 32];  // [row][k]
    __shared__ __align__(16) bf16 lds_w[64 * 32];   // [n][k]
    const int t = threadIdx.x;
    const int w = t >> 5, lane = t & 31;
    const int half = lane >> 4, ln = lane & 15;
    const int rowbase = blockIdx.x * 128;
    const int colbase = blockIdx.y * 64;
    const int sr = w & 3, sc = w >> 2;              // 4 row-groups x 2 col-groups of 32
    v8f a00 = vzero8(), a01 = vzero8(), a10 = vzero8(), a11 = vzero8();

    for (int kk = 0; kk < K; kk += 32) {
        // stage A tile 128x32: 512 uint4 segments, 2 per thread (clamped rows)
#pragma unroll
        for (int i = 0; i < 2; ++i) {
            int id  = t + i * 256;
            int ar  = id >> 2, seg = id & 3;
            int gr  = rowbase + ar; if (gr > M - 1) gr = M - 1;
            uint4 val = *reinterpret_cast<const uint4*>(A + (size_t)gr * K + kk + seg * 8);
            *reinterpret_cast<uint4*>(&lds_a[ar * 32 + seg * 8]) = val;
        }
        // stage W^T tile 64x32: contiguous rows, 1 uint4 per thread
        {
            int n = t >> 2, seg = t & 3;
            uint4 val = *reinterpret_cast<const uint4*>(
                WT + (size_t)(colbase + n) * K + kk + seg * 8);
            *reinterpret_cast<uint4*>(&lds_w[n * 32 + seg * 8]) = val;
        }
        __syncthreads();
        // A fragments (rows sr*32 and sr*32+16)
        const bf16* ab0 = &lds_a[(sr * 32 + ln) * 32];
        const bf16* ab1 = &lds_a[(sr * 32 + 16 + ln) * 32];
        v16bf af0 = cat16(ld8(ab0 + half * 16), ld8(ab0 + half * 16 + 8));
        v16bf af1 = cat16(ld8(ab1 + half * 16), ld8(ab1 + half * 16 + 8));
        // B fragments (cols sc*32 and sc*32+16), contiguous in [n][k]
        const bf16* bb0 = &lds_w[(sc * 32 + ln) * 32 + half * 16];
        const bf16* bb1 = &lds_w[(sc * 32 + 16 + ln) * 32 + half * 16];
        v16bf bf0 = cat16(ld8(bb0), ld8(bb0 + 8));
        v16bf bf1 = cat16(ld8(bb1), ld8(bb1 + 8));
        a00 = WMMA_BF16(af0, bf0, a00);
        a01 = WMMA_BF16(af0, bf1, a01);
        a10 = WMMA_BF16(af1, bf0, a10);
        a11 = WMMA_BF16(af1, bf1, a11);
        __syncthreads();
    }
    const int c0 = colbase + sc * 32 + ln;
    const int c1 = c0 + 16;
    const float bv0 = bias[c0], bv1 = bias[c1];
#pragma unroll
    for (int r = 0; r < 8; ++r) {
        int m  = half * 8 + r;
        int r0 = rowbase + sr * 32 + m;
        int r1 = r0 + 16;
        if (r0 < M) {
            float v0 = a00[r] + bv0, v1 = a01[r] + bv1;
            if (outF) { outF[(size_t)r0 * N + c0] = v0; outF[(size_t)r0 * N + c1] = v1; }
            if (outB) { outB[(size_t)r0 * N + c0] = (bf16)v0; outB[(size_t)r0 * N + c1] = (bf16)v1; }
        }
        if (r1 < M) {
            float v0 = a10[r] + bv0, v1 = a11[r] + bv1;
            if (outF) { outF[(size_t)r1 * N + c0] = v0; outF[(size_t)r1 * N + c1] = v1; }
            if (outB) { outB[(size_t)r1 * N + c0] = (bf16)v0; outB[(size_t)r1 * N + c1] = (bf16)v1; }
        }
    }
}

// ---------------------------------------------------------------------------
// Flash attention: one block = (b,h,128 q rows); 8 waves x 16 q rows each.
// q fragments live in registers for the whole key loop.  K staged row-major;
// V staged from the pre-transposed/interleaved vT with plain b128 copies.
// P->A-layout transpose via per-wave LDS, packed b32 stores (interleave makes
// the two S-tiles' columns adjacent).
__global__ __launch_bounds__(256) void k_attn(
        const bf16* __restrict__ q, const bf16* __restrict__ kmat,
        const bf16* __restrict__ vT, bf16* __restrict__ outB) {
    __shared__ __align__(16) bf16 lds_k [32 * 64];    // [key][d]
    __shared__ __align__(16) bf16 lds_vT[64 * 32];    // [d][pos]
    __shared__ __align__(16) bf16 lds_p [8 * 16 * 32];// per-wave P scratch
    const int t = threadIdx.x, w = t >> 5, lane = t & 31;
    const int half = lane >> 4, ln = lane & 15;
    const int bh = blockIdx.x / 5, qt = blockIdx.x % 5;
    const int b = bh >> 3, h = bh & 7;
    const int qbase = qt * 128 + w * 16;

    // preload q A-fragments (16 rows x 64 d -> two 16x32 fragments)
    int qr = qbase + ln; if (qr > qN - 1) qr = qN - 1;
    const bf16* qp = q + ((size_t)(b * qN + qr)) * Cz + h * 64;
    v16bf qa0 = cat16(ld8(qp + half * 16),      ld8(qp + half * 16 + 8));
    v16bf qa1 = cat16(ld8(qp + 32 + half * 16), ld8(qp + 40 + half * 16));

    const bf16* vbase = vT + ((size_t)(b * Hz + h)) * Dz * kNp;

    v8f O0 = vzero8(), O1 = vzero8(), O2 = vzero8(), O3 = vzero8();
    float mrow[8], lrow[8];
#pragma unroll
    for (int r = 0; r < 8; ++r) { mrow[r] = -3.0e38f; lrow[r] = 0.f; }
    bf16* pw = &lds_p[w * 16 * 32];

    for (int kk = 0; kk < kN; kk += 32) {
        {   // stage K chunk (clamped keys; masked later) and V chunk (padded rows)
            int lk = t >> 3, seg = t & 7;
            int gk = kk + lk; if (gk > kN - 1) gk = kN - 1;
            uint4 kv = *reinterpret_cast<const uint4*>(
                kmat + ((size_t)(b * kN + gk)) * Cz + h * 64 + seg * 8);
            *reinterpret_cast<uint4*>(&lds_k[lk * 64 + seg * 8]) = kv;
            int d = t >> 2, vseg = t & 3;
            uint4 vv = *reinterpret_cast<const uint4*>(
                vbase + (size_t)d * kNp + kk + vseg * 8);
            *reinterpret_cast<uint4*>(&lds_vT[d * 32 + vseg * 8]) = vv;
        }
        __syncthreads();

        // S = q . k^T : two 16x16 S tiles (keys kk..+15, kk+16..+31), K-dim = D = 64
        v8f S0 = vzero8(), S1 = vzero8();
        {
            const bf16* kb = &lds_k[ln * 64 + half * 16];
            v16bf bA = cat16(ld8(kb),      ld8(kb + 8));
            v16bf bB = cat16(ld8(kb + 32), ld8(kb + 40));
            S0 = WMMA_BF16(qa0, bA, S0);
            S0 = WMMA_BF16(qa1, bB, S0);
        }
        {
            const bf16* kb = &lds_k[(16 + ln) * 64 + half * 16];
            v16bf bA = cat16(ld8(kb),      ld8(kb + 8));
            v16bf bB = cat16(ld8(kb + 32), ld8(kb + 40));
            S1 = WMMA_BF16(qa0, bA, S1);
            S1 = WMMA_BF16(qa1, bB, S1);
        }
        // scale + key mask
        int key0 = kk + ln, key1 = kk + 16 + ln;
#pragma unroll
        for (int r = 0; r < 8; ++r) {
            S0[r] = (key0 < kN) ? S0[r] * SCALE : -3.0e38f;
            S1[r] = (key1 < kN) ? S1[r] * SCALE : -3.0e38f;
        }
        // online softmax (row reductions over the 16 lanes of each half)
        float p0[8], p1[8];
#pragma unroll
        for (int r = 0; r < 8; ++r) {
            float rm = fmaxf(S0[r], S1[r]);
            rm = fmaxf(rm, __shfl_xor(rm, 1, 32));
            rm = fmaxf(rm, __shfl_xor(rm, 2, 32));
            rm = fmaxf(rm, __shfl_xor(rm, 4, 32));
            rm = fmaxf(rm, __shfl_xor(rm, 8, 32));
            float mnew = fmaxf(mrow[r], rm);
            float scl  = __expf(mrow[r] - mnew);
            p0[r] = __expf(S0[r] - mnew);
            p1[r] = __expf(S1[r] - mnew);
            float rs = p0[r] + p1[r];
            rs += __shfl_xor(rs, 1, 32);
            rs += __shfl_xor(rs, 2, 32);
            rs += __shfl_xor(rs, 4, 32);
            rs += __shfl_xor(rs, 8, 32);
            lrow[r] = lrow[r] * scl + rs;
            mrow[r] = mnew;
            O0[r] *= scl; O1[r] *= scl; O2[r] *= scl; O3[r] *= scl;
        }
        // P (C-layout) -> LDS in interleaved pos order: pos(p0)=2*ln, pos(p1)=2*ln+1
        unsigned* pw32 = reinterpret_cast<unsigned*>(pw);
#pragma unroll
        for (int r = 0; r < 8; ++r) {
            int m = half * 8 + r;
            pw32[m * 16 + ln] = pack2(p0[r], p1[r]);
        }
        asm volatile("s_wait_dscnt 0x0" ::: "memory");  // wave-internal DS ordering
        const bf16* pb = &pw[ln * 32];
        v16bf pa = cat16(ld8(pb + half * 16), ld8(pb + half * 16 + 8));
        // O += P @ V : four 16x16 d-groups (V rows share the interleaved pos order)
#pragma unroll
        for (int g = 0; g < 4; ++g) {
            const bf16* vb = &lds_vT[(g * 16 + ln) * 32 + half * 16];
            v16bf bv = cat16(ld8(vb), ld8(vb + 8));
            if (g == 0) O0 = WMMA_BF16(pa, bv, O0);
            if (g == 1) O1 = WMMA_BF16(pa, bv, O1);
            if (g == 2) O2 = WMMA_BF16(pa, bv, O2);
            if (g == 3) O3 = WMMA_BF16(pa, bv, O3);
        }
        __syncthreads();
    }
    // normalize + store bf16 (input to the projection GEMM)
#pragma unroll
    for (int r = 0; r < 8; ++r) {
        int m  = half * 8 + r;
        int gq = qt * 128 + w * 16 + m;
        if (gq < qN) {
            float inv = 1.0f / lrow[r];
            bf16* op = outB + ((size_t)(b * qN + gq)) * Cz + h * 64 + ln;
            op[0]  = (bf16)(O0[r] * inv);
            op[16] = (bf16)(O1[r] * inv);
            op[32] = (bf16)(O2[r] * inv);
            op[48] = (bf16)(O3[r] * inv);
        }
    }
}

// ---------------------------------------------------------------------------
// LayerNorm(tf + xp): row 0 of each batch -> txt_cur buffer, rows 1.. -> d_out
__global__ __launch_bounds__(256) void k_ln1(
        const float* __restrict__ tf, const float* __restrict__ xp,
        const float* __restrict__ g, const float* __restrict__ be,
        float* __restrict__ txtcur, float* __restrict__ textOut) {
    __shared__ float red[256];
    int row = blockIdx.x, t = threadIdx.x;
    const float* a  = tf + (size_t)row * Cz;
    const float* bb = xp + (size_t)row * Cz;
    float x0 = a[t] + bb[t], x1 = a[t + 256] + bb[t + 256];
    red[t] = x0 + x1;
    __syncthreads();
    for (int s = 128; s > 0; s >>= 1) { if (t < s) red[t] += red[t + s]; __syncthreads(); }
    float mean = red[0] * (1.0f / Cz);
    __syncthreads();
    float d0 = x0 - mean, d1 = x1 - mean;
    red[t] = d0 * d0 + d1 * d1;
    __syncthreads();
    for (int s = 128; s > 0; s >>= 1) { if (t < s) red[t] += red[t + s]; __syncthreads(); }
    float rstd = rsqrtf(red[0] * (1.0f / Cz) + LN_EPS);
    int b = row / qN, rr = row % qN;
    float y0 = d0 * rstd * g[t] + be[t];
    float y1 = d1 * rstd * g[t + 256] + be[t + 256];
    if (rr == 0) {
        txtcur[b * Cz + t] = y0;
        txtcur[b * Cz + t + 256] = y1;
    } else {
        float* o = textOut + ((size_t)b * Lz + (rr - 1)) * Cz;
        o[t] = y0; o[t + 256] = y1;
    }
}

// ---------------------------------------------------------------------------
// cosine similarity argmax over text rows (sigmoid is monotonic -> skip it)
__global__ __launch_bounds__(256) void k_sim(
        const float* __restrict__ vis, const int* __restrict__ mask,
        const float* __restrict__ textCur, int* __restrict__ idx) {
    __shared__ float svis[Cz];
    __shared__ float red[256];
    __shared__ float bv[8];
    __shared__ int   bi[8];
    int b = blockIdx.x, t = threadIdx.x;
    svis[t] = vis[b * Cz + t];
    svis[t + 256] = vis[b * Cz + t + 256];
    red[t] = svis[t] * svis[t] + svis[t + 256] * svis[t + 256];
    __syncthreads();
    for (int s = 128; s > 0; s >>= 1) { if (t < s) red[t] += red[t + s]; __syncthreads(); }
    float nvis = fmaxf(sqrtf(red[0]), 1e-12f);
    int w = t >> 5, lane = t & 31;
    float bestv = -INFINITY; int besti = 0;
    for (int l = w; l < Lz; l += 8) {
        const float* tc = textCur + ((size_t)b * Lz + l) * Cz;
        float dot = 0.f, nt = 0.f;
#pragma unroll
        for (int j = 0; j < 16; ++j) {
            float tv = tc[lane + 32 * j];
            dot += svis[lane + 32 * j] * tv;
            nt  += tv * tv;
        }
#pragma unroll
        for (int off = 16; off >= 1; off >>= 1) {
            dot += __shfl_xor(dot, off, 32);
            nt  += __shfl_xor(nt,  off, 32);
        }
        float score = mask[b * Lz + l] ? -3.0e38f
                    : dot / (nvis * fmaxf(sqrtf(nt), 1e-12f));
        if (score > bestv) { bestv = score; besti = l; }
    }
    if (lane == 0) { bv[w] = bestv; bi[w] = besti; }
    __syncthreads();
    if (t == 0) {
        float best = bv[0]; int bidx = bi[0];
        for (int i = 1; i < 8; ++i)
            if (bv[i] > best || (bv[i] == best && bi[i] < bidx)) { best = bv[i]; bidx = bi[i]; }
        idx[b] = bidx;
    }
}

// ---------------------------------------------------------------------------
__global__ void k_gather(const float* __restrict__ txtcur, const float* __restrict__ textCur,
                         const int* __restrict__ idx, bf16* __restrict__ txt2) {
    int e = blockIdx.x * 256 + threadIdx.x;
    if (e >= Bz * Cz) return;
    int b = e / Cz, c = e % Cz;
    float v = txtcur[e] + textCur[((size_t)b * Lz + idx[b]) * Cz + c];
    txt2[e] = (bf16)v;
}

// ---------------------------------------------------------------------------
__global__ __launch_bounds__(256) void k_ln2(
        const float* __restrict__ xin, const float* __restrict__ g,
        const float* __restrict__ be, float* __restrict__ out) {
    __shared__ float red[256];
    int row = blockIdx.x, t = threadIdx.x;
    const float* a = xin + (size_t)row * Cz;
    float x0 = a[t], x1 = a[t + 256];
    red[t] = x0 + x1;
    __syncthreads();
    for (int s = 128; s > 0; s >>= 1) { if (t < s) red[t] += red[t + s]; __syncthreads(); }
    float mean = red[0] * (1.0f / Cz);
    __syncthreads();
    float d0 = x0 - mean, d1 = x1 - mean;
    red[t] = d0 * d0 + d1 * d1;
    __syncthreads();
    for (int s = 128; s > 0; s >>= 1) { if (t < s) red[t] += red[t + s]; __syncthreads(); }
    float rstd = rsqrtf(red[0] * (1.0f / Cz) + LN_EPS);
    out[(size_t)row * Cz + t]       = d0 * rstd * g[t] + be[t];
    out[(size_t)row * Cz + t + 256] = d1 * rstd * g[t + 256] + be[t + 256];
}

// ---------------------------------------------------------------------------
extern "C" void kernel_launch(void* const* d_in, const int* in_sizes, int n_in,
                              void* d_out, int out_size, void* d_ws, size_t ws_size,
                              hipStream_t stream) {
    const float* txt_token = (const float*)d_in[0];
    const float* text_token= (const float*)d_in[1];
    const int*   text_mask = (const int*)  d_in[2];
    const float* vis_token = (const float*)d_in[3];
    const float* tmpl_tok  = (const float*)d_in[4];
    const float* tmpl_pos  = (const float*)d_in[5];
    const float* t_pos_w   = (const float*)d_in[6];
    const float* v_pos_w   = (const float*)d_in[7];
    const float* Wq = (const float*)d_in[8];   const float* bq = (const float*)d_in[9];
    const float* Wk = (const float*)d_in[10];  const float* bk = (const float*)d_in[11];
    const float* Wv = (const float*)d_in[12];  const float* bvv= (const float*)d_in[13];
    const float* Wp = (const float*)d_in[14];  const float* bp = (const float*)d_in[15];
    const float* Wt = (const float*)d_in[16];  const float* bt = (const float*)d_in[17];
    const float* g1 = (const float*)d_in[18];  const float* be1= (const float*)d_in[19];
    const float* g2 = (const float*)d_in[20];  const float* be2= (const float*)d_in[21];

    char* ws = (char*)d_ws;
    float* tfF     = (float*)(ws + OFF_TFF);
    bf16*  tfB     = (bf16*) (ws + OFF_P1);     // then reused as attn-out bf16
    bf16*  attnB   = (bf16*) (ws + OFF_P1);
    bf16*  kinB    = (bf16*) (ws + OFF_P2);     // then reused as xp f32
    float* xpF     = (float*)(ws + OFF_P2);
    bf16*  vinB    = (bf16*) (ws + OFF_VIN);
    bf16*  wbf     = (bf16*) (ws + OFF_WBF);
    bf16*  qB      = (bf16*) (ws + OFF_Q);
    bf16*  kB      = (bf16*) (ws + OFF_K);
    bf16*  vB      = (bf16*) (ws + OFF_V);
    bf16*  vTB     = (bf16*) (ws + OFF_VT);
    float* xembed  = (float*)(ws + OFF_XE);
    float* txtcur  = (float*)(ws + OFF_TXTC);
    bf16*  txt2B   = (bf16*) (ws + OFF_TXT2);
    float* txtproj = (float*)(ws + OFF_TXTP);
    int*   idxbuf  = (int*)  (ws + OFF_IDX);

    float* outp     = (float*)d_out;            // txt_cur: first B*C floats
    float* text_out = outp + (size_t)Bz * Cz;   // text_cur: B*L*C floats

    constexpr int WN = Cz * Cz;
    bf16* wqB = wbf;            bf16* wkB = wbf + WN;
    bf16* wvB = wbf + 2 * WN;   bf16* wpB = wbf + 3 * WN;
    bf16* wtB = wbf + 4 * WN;

    // weights -> bf16, transposed to [N][K]
    k_convT<<<WN / 256, 256, 0, stream>>>(Wq, wqB);
    k_convT<<<WN / 256, 256, 0, stream>>>(Wk, wkB);
    k_convT<<<WN / 256, 256, 0, stream>>>(Wv, wvB);
    k_convT<<<WN / 256, 256, 0, stream>>>(Wp, wpB);
    k_convT<<<WN / 256, 256, 0, stream>>>(Wt, wtB);

    // positional scan + input assembly
    k_scan<<<Bz, Lz, 0, stream>>>(text_mask, xembed);
    k_fill_tf<<<(int)(N_TFE / 256), 256, 0, stream>>>(txt_token, text_token, t_pos_w,
                                                      xembed, tfF, tfB);
    k_fill_kv<<<(int)(N_KVE / 256), 256, 0, stream>>>(vis_token, v_pos_w, tmpl_tok,
                                                      tmpl_pos, kinB, vinB);

    // Q/K/V projections (WMMA bf16)
    const int Mq = Bz * qN;   // 8208
    const int Mk = Bz * kN;   // 16400
    k_gemm_bf16<<<dim3((Mq + 127) / 128, Cz / 64), 256, 0, stream>>>(tfB,  wqB, bq,  nullptr, qB, Mq);
    k_gemm_bf16<<<dim3((Mk + 127) / 128, Cz / 64), 256, 0, stream>>>(kinB, wkB, bk,  nullptr, kB, Mk);
    k_gemm_bf16<<<dim3((Mk + 127) / 128, Cz / 64), 256, 0, stream>>>(vinB, wvB, bvv, nullptr, vB, Mk);

    // V -> transposed/interleaved layout for attention staging
    k_vT<<<(int)((N_VT + 255) / 256), 256, 0, stream>>>(vB, vTB);

    // flash attention (writes attnB, reusing tfB's slot)
    k_attn<<<Bz * Hz * 5, 256, 0, stream>>>(qB, kB, vTB, attnB);

    // output projection (xpF reuses kinB's slot)
    k_gemm_bf16<<<dim3((Mq + 127) / 128, Cz / 64), 256, 0, stream>>>(attnB, wpB, bp, xpF, nullptr, Mq);

    // residual + LN1 -> txt_cur buffer + text_cur (directly into d_out)
    k_ln1<<<Mq, 256, 0, stream>>>(tfF, xpF, g1, be1, txtcur, text_out);

    // similarity argmax, gather, final GEMM + LN2
    k_sim<<<Bz, 256, 0, stream>>>(vis_token, text_mask, text_out, idxbuf);
    k_gather<<<(Bz * Cz) / 256, 256, 0, stream>>>(txtcur, text_out, idxbuf, txt2B);
    k_gemm_bf16<<<dim3(1, Cz / 64), 256, 0, stream>>>(txt2B, wtB, bt, txtproj, nullptr, Bz);
    k_ln2<<<Bz, 256, 0, stream>>>(txtproj, g2, be2, outp);
}
